// GATLayer_87814901334398
// MI455X (gfx1250) — matrix operand order repaired and checked by reading, not compiled, not run on hardware
//
#include <hip/hip_runtime.h>
#include <hip/hip_bf16.h>

typedef __attribute__((ext_vector_type(16))) _Float16 v16h;
typedef __attribute__((ext_vector_type(8)))  float    v8f;

#define ALPHA   0.2f
#define NEG_INF -9e15f
#define NN      8192
#define FD      128
#define MWORDS  (NN / 32)   // 256 mask words per row

// ---------------------------------------------------------------------------
// Kernel 1: Wh = h @ W (fp32), store WhT as f16 (for WMMA B operand),
//           e_src[i] = Wh_i . a[0:128], e_dst[i] = Wh_i . a[128:256]
// ---------------------------------------------------------------------------
__global__ void __launch_bounds__(128)
wh_kernel(const float* __restrict__ h, const float* __restrict__ W,
          const float* __restrict__ a, _Float16* __restrict__ WhT,
          float* __restrict__ e_src, float* __restrict__ e_dst)
{
    __shared__ float hs[FD];
    __shared__ float r1[FD];
    __shared__ float r2[FD];
    const int i = blockIdx.x;
    const int t = threadIdx.x;
    hs[t] = h[i * FD + t];
    __syncthreads();
    float acc = 0.f;
#pragma unroll 8
    for (int k = 0; k < FD; ++k) acc = fmaf(hs[k], W[k * FD + t], acc);
    WhT[(size_t)t * NN + i] = (_Float16)acc;   // transposed f16 copy
    r1[t] = acc * a[t];
    r2[t] = acc * a[FD + t];
    __syncthreads();
    for (int s = 64; s > 0; s >>= 1) {
        if (t < s) { r1[t] += r1[t + s]; r2[t] += r2[t + s]; }
        __syncthreads();
    }
    if (t == 0) { e_src[i] = r1[0]; e_dst[i] = r2[0]; }
}

// ---------------------------------------------------------------------------
// Kernel 2: single pass over adj (the 256MB roofline term).
// Packs adj row -> bitmask, computes row max m_i and denom_i = sum exp(e-m).
// ---------------------------------------------------------------------------
__global__ void __launch_bounds__(256)
stats_kernel(const int* __restrict__ adj, const float* __restrict__ e_src,
             const float* __restrict__ e_dst, unsigned* __restrict__ mask,
             float* __restrict__ rowmax, float* __restrict__ denom)
{
    __shared__ float red[256];
    const int i = blockIdx.x;
    const int t = threadIdx.x;
    const float es = e_src[i];
    const int4*   arow4 = (const int4*)(adj + (size_t)i * NN + t * 32);
    const float4* ed4   = (const float4*)(e_dst + t * 32);

    int   av[32];
    float ed[32];
#pragma unroll
    for (int q = 0; q < 8; ++q) {
        const int4 w = arow4[q];
        av[4*q+0] = w.x; av[4*q+1] = w.y; av[4*q+2] = w.z; av[4*q+3] = w.w;
        const float4 f = ed4[q];
        ed[4*q+0] = f.x; ed[4*q+1] = f.y; ed[4*q+2] = f.z; ed[4*q+3] = f.w;
    }

    float ev[32];
    unsigned mw = 0u;
#pragma unroll
    for (int c = 0; c < 32; ++c) {
        const float v  = es + ed[c];
        const float lr = v > 0.f ? v : ALPHA * v;
        const bool on  = av[c] > 0;
        mw |= (on ? 1u : 0u) << c;
        ev[c] = on ? lr : NEG_INF;
    }
    mask[(size_t)i * MWORDS + t] = mw;

    float lmax = NEG_INF;
#pragma unroll
    for (int c = 0; c < 32; ++c) lmax = fmaxf(lmax, ev[c]);
    red[t] = lmax;
    __syncthreads();
    for (int s = 128; s > 0; s >>= 1) {
        if (t < s) red[t] = fmaxf(red[t], red[t + s]);
        __syncthreads();
    }
    const float m = red[0];
    __syncthreads();

    float lsum = 0.f;
#pragma unroll
    for (int c = 0; c < 32; ++c) lsum += __expf(ev[c] - m);
    red[t] = lsum;
    __syncthreads();
    for (int s = 128; s > 0; s >>= 1) {
        if (t < s) red[t] += red[t + s];
        __syncthreads();
    }
    if (t == 0) { rowmax[i] = m; denom[i] = red[0]; }
}

// ---------------------------------------------------------------------------
// Kernel 3: fused P @ Wh via v_wmma_f32_16x16x32_f16, 32 rows per block.
// 8 waves = 2 column-groups (4 ct tiles) x 4-way split-K. Each wave holds TWO
// A-fragments (row-tiles i0..i0+15 and i0+16..i0+31) and issues 2 WMMAs per
// B-tile load -> halves B (WhT) L2 traffic vs 16-row blocks.
// Per step: A-feeding loads + 8 b128 B loads, sched_barrier pins them above
// the 32-deep v_exp_f32 chain (latency hiding), then 8 WMMAs back-to-back.
//
// A layout (16-bit 16x32): lane=(m=lane&15, hi=lane>>4);
//   elem e -> K=hi*16+e
// B layout (32x16): lane=(n=m, bhi=hi); elem e -> K=bhi*16+e  (32B v16h load)
// C layout: vgpr r -> row = hi*8+r (within row-tile), col = n.
// ---------------------------------------------------------------------------
__global__ void __launch_bounds__(256)
av_kernel(const _Float16* __restrict__ WhT, const unsigned* __restrict__ mask,
          const float* __restrict__ e_src, const float* __restrict__ e_dst,
          const float* __restrict__ rowmax, const float* __restrict__ denom,
          float* __restrict__ out)
{
    __shared__ float cacc[32 * FD];          // 16 KB
    const int tid  = threadIdx.x;
    const int wave = tid >> 5;
    const int ctg  = wave & 1;               // column group: cts [4*ctg, 4*ctg+4)
    const int ks   = wave >> 1;              // split-K id 0..3
    const int lane = tid & 31;
    const int m    = lane & 15;
    const int hi   = lane >> 4;
    const int i0   = blockIdx.x * 32;

    for (int t = tid; t < 32 * FD; t += 256) cacc[t] = 0.f;

    const float es0 = e_src[i0 + m];
    const float es1 = e_src[i0 + 16 + m];
    const float mi0 = rowmax[i0 + m];
    const float mi1 = rowmax[i0 + 16 + m];
    const unsigned* mrow0 = mask + (size_t)(i0 + m) * MWORDS;
    const unsigned* mrow1 = mask + (size_t)(i0 + 16 + m) * MWORDS;
    const _Float16* wbase = WhT + (size_t)(ctg * 64 + m) * NN + hi * 16;

    const v8f zero = {};
    v8f acc0[4], acc1[4];                    // [ct] for row-tile 0 / 1
#pragma unroll
    for (int c = 0; c < 4; ++c) { acc0[c] = zero; acc1[c] = zero; }

    for (int s = ks; s < MWORDS; s += 4) {
        const int j0 = s * 32;

        // 1) A-feeding loads (mask words + e_dst, all L2-resident)
        const unsigned mw0 = mrow0[s];
        const unsigned mw1 = mrow1[s];
        const float4* edlo4 = (const float4*)(e_dst + j0 + hi * 16);
        const float4* edhi4 = (const float4*)(e_dst + j0 + hi * 16 + 8);
        const float4 l0 = edlo4[0], l1 = edlo4[1];
        const float4 h0 = edhi4[0], h1 = edhi4[1];

        // 2) 4 B tiles (8 x global_load_b128), pinned before the VALU chain
        v16h bfrag[4];
#pragma unroll
        for (int c = 0; c < 4; ++c)
            bfrag[c] = *(const v16h*)(wbase + (size_t)c * (16 * NN) + j0);

        __builtin_amdgcn_sched_barrier(0);   // loads stay above, ALU below

        // 3) 32-deep exp/LeakyReLU chain hides the B-tile latency
        float edv[16];
        edv[0]=l0.x; edv[1]=l0.y; edv[2]=l0.z;  edv[3]=l0.w;
        edv[4]=l1.x; edv[5]=l1.y; edv[6]=l1.z;  edv[7]=l1.w;
        edv[8]=h0.x; edv[9]=h0.y; edv[10]=h0.z; edv[11]=h0.w;
        edv[12]=h1.x; edv[13]=h1.y; edv[14]=h1.z; edv[15]=h1.w;

        v16h afrag0, afrag1;
#pragma unroll
        for (int e = 0; e < 16; ++e) {
            const int k = hi * 16 + e;
            const float v0 = es0 + edv[e];
            const float v1 = es1 + edv[e];
            const float r0 = v0 > 0.f ? v0 : ALPHA * v0;
            const float r1 = v1 > 0.f ? v1 : ALPHA * v1;
            const float e0 = ((mw0 >> k) & 1u) ? r0 : NEG_INF;
            const float e1 = ((mw1 >> k) & 1u) ? r1 : NEG_INF;
            afrag0[e] = (_Float16)__expf(e0 - mi0);
            afrag1[e] = (_Float16)__expf(e1 - mi1);
        }

        // 4) 8 WMMAs back-to-back; each B tile feeds both row-tiles
#pragma unroll
        for (int c = 0; c < 4; ++c) {
            acc0[c] = __builtin_amdgcn_wmma_f32_16x16x32_f16(
                false, afrag0, false, bfrag[c], (short)0, acc0[c], false, false);
            acc1[c] = __builtin_amdgcn_wmma_f32_16x16x32_f16(
                false, afrag1, false, bfrag[c], (short)0, acc1[c], false, false);
        }
    }

    __syncthreads();   // all waves' cacc init done; now reduce
#pragma unroll
    for (int c = 0; c < 4; ++c) {
        const int col = (ctg * 4 + c) * 16 + m;
#pragma unroll
        for (int r = 0; r < 8; ++r) {
            const int row = hi * 8 + r;
            atomicAdd(&cacc[row * FD + col], acc0[c][r]);
            atomicAdd(&cacc[(16 + row) * FD + col], acc1[c][r]);
        }
    }
    __syncthreads();

    for (int t = tid; t < 32 * FD; t += 256) {
        const int row = t >> 7;
        const int col = t & (FD - 1);
        out[(size_t)(i0 + row) * FD + col] = cacc[t] / denom[i0 + row];
    }
}

// ---------------------------------------------------------------------------
extern "C" void kernel_launch(void* const* d_in, const int* in_sizes, int n_in,
                              void* d_out, int out_size, void* d_ws, size_t ws_size,
                              hipStream_t stream)
{
    const float* h   = (const float*)d_in[0];
    const int*   adj = (const int*)d_in[1];
    const float* W   = (const float*)d_in[2];
    const float* a   = (const float*)d_in[3];
    float* out = (float*)d_out;

    char* ws = (char*)d_ws;
    auto align256 = [](size_t x) { return (x + 255) & ~(size_t)255; };

    _Float16* WhT   = (_Float16*)ws; ws += align256((size_t)FD * NN * sizeof(_Float16));
    float*    es    = (float*)ws;    ws += align256((size_t)NN * sizeof(float));
    float*    ed    = (float*)ws;    ws += align256((size_t)NN * sizeof(float));
    unsigned* mask  = (unsigned*)ws; ws += align256((size_t)NN * MWORDS * sizeof(unsigned));
    float*    rmax  = (float*)ws;    ws += align256((size_t)NN * sizeof(float));
    float*    den   = (float*)ws;    ws += align256((size_t)NN * sizeof(float));

    wh_kernel<<<NN, 128, 0, stream>>>(h, W, a, WhT, es, ed);
    stats_kernel<<<NN, 256, 0, stream>>>(adj, es, ed, mask, rmax, den);
    av_kernel<<<NN / 32, 256, 0, stream>>>(WhT, mask, es, ed, rmax, den, out);
}
